// PCConv_28501402976578
// MI455X (gfx1250) — matrix operand order repaired and checked by reading, not compiled, not run on hardware
//
#include <hip/hip_runtime.h>
#include <cstdint>
#include <cstddef>

// ---------------------------------------------------------------------------
// PCConv for MI455X (gfx1250, wave32, WMMA)
//
// Shapes (hardcoded from setup_inputs):
//   B=4, N=8000, QIN=90, QOUT=60, CIN=COUT=32, H1=32, H2=64
//
// Roofline: must stream f_in (369 MB fp32) + write out (246 MB fp32)
//   -> ~26 us floor at 23.3 TB/s. Total math ~15 GFLOP -> ~1-2 us on the
//   f16 WMMA path. Design: stream f_in once, run both contractions on
//   v_wmma_f32_16x16x32_f16 (f16 in / f32 accumulate), never spill agg to HBM.
//
// Pipeline:
//   K0: pack Wf (32x32) into WMMA B-fragment layout (2 co-tiles).
//   K1: per (b,o): sphere dists, antipodal min, rank (double-argsort),
//       mask, WeightNet MLP -> mw = mask*w, packed into WMMA B fragments
//       over the i (K) dimension, 90 padded to 96 = 3 chunks of K=32.
//       (1.5 MB workspace -> L2-resident for all 2000 tile blocks.)
//   K2: per (b, n-tile of 16): stream f_in tile -> f16 in LDS [c][n][i96],
//       stage A: agg[n,o,c] = sum_i f*mw, channels processed in PAIRS
//       (2 independent WMMA chains, packed b32 agg spills),
//       stage B: out[n,o,co] = agg @ Wf + bias, two o per iteration
//       (4 independent WMMA chains).
// ---------------------------------------------------------------------------

typedef _Float16 v16h __attribute__((ext_vector_type(16)));
typedef float    v8f  __attribute__((ext_vector_type(8)));

#define NB    4
#define NN    8000
#define QIN   90
#define QOUT  60
#define CIN   32
#define COUT  32
#define HH1   32
#define HH2   64
#define KMAXS 16
#define DMAXS 1.0f
#define EPSC  1e-7f

#define NTILE (NN / 16)        // 500 n-tiles per batch
#define IPAD  96               // qin padded to 3 K-chunks of 32
#define NQCH  3                // K chunks per stage-A contraction

// workspace layout (bytes)
#define MWB_BYTES ((size_t)NB * 4 * NQCH * CIN * 32 * 32)  // 1,572,864
#define WFB_DW_OFF (MWB_BYTES / 4)                         // dword offset of WfB

union Frag { uint4 u[2]; v16h h; };

__device__ __forceinline__ unsigned pack2h(float a, float b) {
  unsigned short ua = __builtin_bit_cast(unsigned short, (_Float16)a);
  unsigned short ub = __builtin_bit_cast(unsigned short, (_Float16)b);
  return (unsigned)ua | ((unsigned)ub << 16);
}
__device__ __forceinline__ unsigned short h1b(float a) {
  return __builtin_bit_cast(unsigned short, (_Float16)a);
}

// ---------------------------------------------------------------------------
// K0: Wf -> B-fragment layout.
// Dense 16-bit B 32x16 per-lane layout (extrapolated from ISA §7.12.2/7.12.4):
//   lane l, dword v (0..7): K = (v<4 ? 0 : 16) + (l<16 ? 0 : 8) + 2(v&3) + {lo,hi half}, col = l%16
// ---------------------------------------------------------------------------
__global__ void pcconv_wfb_kernel(const float* __restrict__ Wf,
                                  unsigned* __restrict__ WfB) {
  for (int wi = threadIdx.x; wi < 512; wi += blockDim.x) {
    int v    = wi & 7;
    int lane = (wi >> 3) & 31;
    int cot  = wi >> 8;                 // co-tile 0..1
    int K    = ((v & 4) ? 16 : 0) + ((lane & 16) ? 8 : 0) + 2 * (v & 3);   // c (contraction) index
    int co   = cot * 16 + (lane & 15);
    WfB[cot * 256 + lane * 8 + v] =
        pack2h(Wf[K * COUT + co], Wf[(K + 1) * COUT + co]);
  }
}

// ---------------------------------------------------------------------------
// K1: one block per (b,o). 128 threads; threads 0..89 own one q_in each.
// ---------------------------------------------------------------------------
__global__ __launch_bounds__(128) void pcconv_weight_kernel(
    const float* __restrict__ ang_in, const float* __restrict__ ang_out,
    const float* __restrict__ W1, const float* __restrict__ b1,
    const float* __restrict__ W2, const float* __restrict__ b2,
    const float* __restrict__ W3, const float* __restrict__ b3,
    unsigned* __restrict__ mwB) {
  __shared__ float sd[QIN];
  __shared__ float swm[QIN][CIN];   // masked WeightNet output

  const int blk = blockIdx.x;
  const int b = blk / QOUT;
  const int o = blk % QOUT;
  const int t = threadIdx.x;

  // ang_out unit vector + norm (every thread, 3 cached loads)
  const float ox = ang_out[(b * QOUT + o) * 3 + 0];
  const float oy = ang_out[(b * QOUT + o) * 3 + 1];
  const float oz = ang_out[(b * QOUT + o) * 3 + 2];
  const float on = sqrtf(ox * ox + oy * oy + oz * oz);
  const float oi = (on > 0.f) ? 1.f / on : 0.f;
  const float ux = ox * oi, uy = oy * oi, uz = oz * oi;

  float bdv = 0.f;
  if (t < QIN) {
    const float x = ang_in[(b * QIN + t) * 3 + 0];
    const float y = ang_in[(b * QIN + t) * 3 + 1];
    const float z = ang_in[(b * QIN + t) * 3 + 2];
    const float nn = sqrtf(x * x + y * y + z * z);
    const float iv = (nn > 0.f) ? 1.f / nn : 0.f;
    const float dot = (x * ux + y * uy + z * uz) * iv;
    // antipodal min: min(acos(clip(dot)),acos(clip(-dot))) = acos(min(|dot|,1-eps))
    sd[t] = acosf(fminf(fabsf(dot), 1.f - EPSC));
    bdv = on - nn;                     // bval_diff
  }
  __syncthreads();

  if (t < QIN) {
    const float dt = sd[t];
    int r = 0;                         // stable double-argsort rank
    for (int j = 0; j < QIN; ++j) {
      const float dj = sd[j];
      r += (dj < dt) || (dj == dt && j < t);
    }
    const bool m = (dt <= DMAXS) && (r < KMAXS);

    // WeightNet: p = [0,0,0,dist,bval_diff]
    float h1[HH1];
#pragma unroll
    for (int h = 0; h < HH1; ++h) {
      float s = dt * W1[3 * HH1 + h] + bdv * W1[4 * HH1 + h] + b1[h];
      h1[h] = fmaxf(s, 0.f);
    }
    float h2[HH2];
    for (int j = 0; j < HH2; ++j) {
      float s = b2[j];
#pragma unroll
      for (int h = 0; h < HH1; ++h) s += h1[h] * W2[h * HH2 + j];
      h2[j] = fmaxf(s, 0.f);
    }
    for (int c = 0; c < CIN; ++c) {
      float s = b3[c];
#pragma unroll
      for (int j = 0; j < HH2; ++j) s += h2[j] * W3[j * CIN + c];
      swm[t][c] = m ? s : 0.f;
    }
  }
  __syncthreads();

  // Pack this o's column into the B fragments of its o-tile.
  const int col = o & 15;
  const int ot = o >> 4;
  for (int wi = t; wi < 1536; wi += 128) {     // 3q * 32c * 2 lane-grp * 8 dw
    const int v = wi & 7;
    const int lg = (wi >> 3) & 1;
    const int c = (wi >> 4) & 31;
    const int q = wi >> 9;
    const int lane = col + (lg ? 16 : 0);
    const int i0 = 32 * q + ((v & 4) ? 16 : 0) + (lg ? 8 : 0) + 2 * (v & 3);
    const float a0 = (i0 < QIN) ? swm[i0][c] : 0.f;
    const float a1 = (i0 + 1 < QIN) ? swm[i0 + 1][c] : 0.f;
    mwB[(size_t)(((b * 4 + ot) * NQCH + q) * CIN + c) * 256 + lane * 8 + v] =
        pack2h(a0, a1);
  }
}

// ---------------------------------------------------------------------------
// K2: heavy streaming kernel. One block per (b, n-tile). 4 waves = 4 o-tiles.
// Dynamic LDS: fA [32c][16n][96i] f16 (96KB) + agg [4 waves][16o][16n][32c]
// f16 (64KB) = 160KB -> 2 workgroups per 320KB WGP.
// ---------------------------------------------------------------------------
__global__ __launch_bounds__(128) void pcconv_conv_kernel(
    const float* __restrict__ f_in, const float* __restrict__ bias_out,
    const unsigned* __restrict__ mwB, const unsigned* __restrict__ WfB,
    float* __restrict__ out) {
  extern __shared__ unsigned short smem[];
  unsigned short* fA = smem;                            // [c][n][IPAD]

  const int blk = blockIdx.x;
  const int b = blk / NTILE;
  const int n0 = (blk % NTILE) * 16;
  const int t = threadIdx.x;

  // ---- load f-tile (16 x 90 x 32 fp32), convert to f16, scatter to LDS ----
  const float4* src =
      (const float4*)(f_in + (size_t)(b * NN + n0) * (QIN * CIN));
  for (int e4 = t; e4 < 16 * QIN * CIN / 4; e4 += 128) {   // 90 iters/thread
    const float4 v = src[e4];
    const int idx = e4 * 4;
    const int n = idx / (QIN * CIN);
    const int i = (idx / CIN) % QIN;
    const int c = idx & (CIN - 1);
    fA[((c + 0) * 16 + n) * IPAD + i] = h1b(v.x);
    fA[((c + 1) * 16 + n) * IPAD + i] = h1b(v.y);
    fA[((c + 2) * 16 + n) * IPAD + i] = h1b(v.z);
    fA[((c + 3) * 16 + n) * IPAD + i] = h1b(v.w);
  }
  // zero the i = 90..95 K-padding (keep NaN out of the matrix path)
  for (int e = t; e < CIN * 16 * 6; e += 128) {
    const int c = e / 96;
    const int rem = e % 96;
    fA[(c * 16 + rem / 6) * IPAD + 90 + (rem % 6)] = 0;
  }
  __syncthreads();

  const int wave = t >> 5;
  const int lane = t & 31;
  const int ot = wave;                              // o-tile for this wave
  unsigned short* aggW = smem + CIN * 16 * IPAD + wave * (16 * 16 * CIN);
  unsigned* aggW32 = (unsigned*)aggW;               // dword view, c-pairs

  const int nrow = lane & 15;                       // A-matrix row (M)
  const int kb = (lane & 16) ? 8 : 0;               // A K-chunk base per lane grp
  const int nbase = (lane & 16) ? 8 : 0;            // C-matrix M base per lane grp
  const int ocol = lane & 15;                       // C-matrix N column

  // ---- stage A: agg[n,o,c] = sum_i f[n,i,c] * mw[i,o,c], channel pairs ----
  for (int c = 0; c < CIN; c += 2) {
    v8f acc0 = {}, acc1 = {};
#pragma unroll
    for (int q = 0; q < NQCH; ++q) {
      // 16-bit A 16x32: lane grp0 K[0..8)+[16..24), grp1 K[8..16)+[24..32)
      Frag a0, a1;
      const unsigned short* ap0 = &fA[(c * 16 + nrow) * IPAD + q * 32 + kb];
      const unsigned short* ap1 = ap0 + 16 * IPAD;   // channel c+1
      a0.u[0] = *(const uint4*)(ap0);
      a0.u[1] = *(const uint4*)(ap0 + 16);
      a1.u[0] = *(const uint4*)(ap1);
      a1.u[1] = *(const uint4*)(ap1 + 16);
      // pre-packed masked-weight fragments (L2-resident)
      Frag bm0, bm1;
      const uint4* bp = (const uint4*)(mwB +
          (size_t)(((b * 4 + ot) * NQCH + q) * CIN + c) * 256 + lane * 8);
      bm0.u[0] = bp[0];
      bm0.u[1] = bp[1];
      bm1.u[0] = bp[64];                             // +256 dwords = next c
      bm1.u[1] = bp[65];
      acc0 = __builtin_amdgcn_wmma_f32_16x16x32_f16(
          false, a0.h, false, bm0.h, (short)0, acc0, false, false);
      acc1 = __builtin_amdgcn_wmma_f32_16x16x32_f16(
          false, a1.h, false, bm1.h, (short)0, acc1, false, false);
    }
    // spill C-tiles (M=n, N=o_local) packed as (c,c+1) dwords, b32 stores
#pragma unroll
    for (int r = 0; r < 8; ++r)
      aggW32[((ocol * 16 + nbase + r) * CIN + c) >> 1] = pack2h(acc0[r], acc1[r]);
  }

  // ---- stage B: out[n,o,co] = agg[n,o,:] @ Wf + bias, two o per iter ----
  Frag wf0, wf1;
  {
    const uint4* p0 = (const uint4*)(WfB + lane * 8);
    const uint4* p1 = (const uint4*)(WfB + 256 + lane * 8);
    wf0.u[0] = p0[0]; wf0.u[1] = p0[1];
    wf1.u[0] = p1[0]; wf1.u[1] = p1[1];
  }
  const float bs0 = bias_out[lane & 15];
  const float bs1 = bias_out[16 + (lane & 15)];
  const int olim = (ot == 3) ? 12 : 16;             // o = 60..63 are padding

  for (int ol = 0; ol < olim; ol += 2) {            // wave-uniform loop
    Frag a0, a1;
    const unsigned short* ap0 = &aggW[(ol * 16 + nrow) * CIN + kb];
    const unsigned short* ap1 = ap0 + 16 * CIN;     // o_local + 1
    a0.u[0] = *(const uint4*)(ap0);
    a0.u[1] = *(const uint4*)(ap0 + 16);
    a1.u[0] = *(const uint4*)(ap1);
    a1.u[1] = *(const uint4*)(ap1 + 16);
    v8f c00 = {}, c01 = {}, c10 = {}, c11 = {};
    c00 = __builtin_amdgcn_wmma_f32_16x16x32_f16(
        false, a0.h, false, wf0.h, (short)0, c00, false, false);
    c01 = __builtin_amdgcn_wmma_f32_16x16x32_f16(
        false, a0.h, false, wf1.h, (short)0, c01, false, false);
    c10 = __builtin_amdgcn_wmma_f32_16x16x32_f16(
        false, a1.h, false, wf0.h, (short)0, c10, false, false);
    c11 = __builtin_amdgcn_wmma_f32_16x16x32_f16(
        false, a1.h, false, wf1.h, (short)0, c11, false, false);
    const int o = ot * 16 + ol;
    float* op0 = out + ((size_t)(b * NN + n0) * QOUT + o) * COUT;
    float* op1 = op0 + COUT;                        // o + 1
#pragma unroll
    for (int r = 0; r < 8; ++r) {
      const size_t row = (size_t)(nbase + r) * QOUT * COUT;
      op0[row + ocol]      = c00[r] + bs0;
      op0[row + ocol + 16] = c01[r] + bs1;
      op1[row + ocol]      = c10[r] + bs0;
      op1[row + ocol + 16] = c11[r] + bs1;
    }
  }
}

// ---------------------------------------------------------------------------
extern "C" void kernel_launch(void* const* d_in, const int* in_sizes, int n_in,
                              void* d_out, int out_size, void* d_ws,
                              size_t ws_size, hipStream_t stream) {
  (void)in_sizes; (void)n_in; (void)out_size; (void)ws_size;
  const float* ang_in   = (const float*)d_in[0];
  const float* ang_out  = (const float*)d_in[1];
  const float* f_in     = (const float*)d_in[2];
  const float* W1       = (const float*)d_in[3];
  const float* b1       = (const float*)d_in[4];
  const float* W2       = (const float*)d_in[5];
  const float* b2       = (const float*)d_in[6];
  const float* W3       = (const float*)d_in[7];
  const float* b3       = (const float*)d_in[8];
  const float* Wf       = (const float*)d_in[9];
  const float* bias_out = (const float*)d_in[10];
  float* out = (float*)d_out;

  unsigned* mwB = (unsigned*)d_ws;
  unsigned* WfB = (unsigned*)d_ws + WFB_DW_OFF;

  const int dynLds = (CIN * 16 * IPAD + 4 * 16 * 16 * CIN) * 2;  // 163,840 B
  (void)hipFuncSetAttribute((const void*)pcconv_conv_kernel,
                            hipFuncAttributeMaxDynamicSharedMemorySize, dynLds);

  pcconv_wfb_kernel<<<1, 128, 0, stream>>>(Wf, WfB);
  pcconv_weight_kernel<<<NB * QOUT, 128, 0, stream>>>(
      ang_in, ang_out, W1, b1, W2, b2, W3, b3, mwB);
  pcconv_conv_kernel<<<NB * NTILE, 128, dynLds, stream>>>(
      f_in, bias_out, mwB, WfB, out);
}
